// PFNet_29008209117479
// MI455X (gfx1250) — matrix operand order repaired and checked by reading, not compiled, not run on hardware
//
#include <hip/hip_runtime.h>

typedef __bf16 bf16_t;
typedef __attribute__((ext_vector_type(16))) __bf16 v16bf;
typedef __attribute__((ext_vector_type(8)))  __bf16 v8bf;
typedef __attribute__((ext_vector_type(8)))  float  v8f;

#define BATCH 8
#define NPTS  5000
#define ROWS  (BATCH*NPTS)   // 40000
#define DH    256
#define KNBR  16
#define NBINS 50
#define BINSZ 100

#define BM 64
#define BN 128
#define BK 32

__device__ __forceinline__ float seluf(float x){
  const float scale = 1.0507009873554805f, alpha = 1.6732632423543772f;
  return x > 0.f ? scale * x : scale * alpha * (expf(x) - 1.f);
}

// A fragment: 16-bit A 16x32 layout. lane<16 holds K {0..7, 16..23}; lane>=16 holds {8..15, 24..31}
__device__ __forceinline__ v16bf fragA(const bf16_t* base, int half){
  v8bf lo = *(const v8bf*)(base + 8*half);
  v8bf hi = *(const v8bf*)(base + 16 + 8*half);
  v16bf r;
  #pragma unroll
  for (int i = 0; i < 8; ++i){ r[i] = lo[i]; r[8+i] = hi[i]; }
  return r;
}
// B fragment: lane<16 holds K {0..7, 16..23}; lane>=16 holds {8..15, 24..31}
__device__ __forceinline__ v16bf fragB(const bf16_t* base, int half){
  v8bf lo = *(const v8bf*)(base + 8*half);
  v8bf hi = *(const v8bf*)(base + 16 + 8*half);
  v16bf r;
  #pragma unroll
  for (int i = 0; i < 8; ++i){ r[i] = lo[i]; r[8+i] = hi[i]; }
  return r;
}

__device__ __forceinline__ v8f wmma_bf16(v16bf a, v16bf b, v8f c){
  return __builtin_amdgcn_wmma_f32_16x16x32_bf16(false, a, false, b, (short)0, c, false, false);
}

// ---------------- input encoding: one-hot(12) + 14 props -> bf16 [ROWS x 32] (K padded) ----
__global__ void k_encode(const float* __restrict__ X, bf16_t* __restrict__ enc){
  int idx = blockIdx.x*blockDim.x + threadIdx.x;
  if (idx >= ROWS*32) return;
  int r = idx >> 5, c = idx & 31;
  const float* xr = X + (size_t)r*15;
  float v = 0.f;
  if (c < 12)      { int id = (int)xr[0]; v = (c == id) ? 1.f : 0.f; }
  else if (c < 26) { v = xr[1 + (c - 12)]; }
  enc[idx] = (bf16_t)v;
}

// ---------------- weight pack TRANSPOSED: f32 [K x N] -> bf16 [(rowOff+n) x Kpad] ----------
__global__ void k_packT(const float* __restrict__ src, bf16_t* __restrict__ dst,
                        int K, int N, int Kpad, int rowOff){
  int idx = blockIdx.x*blockDim.x + threadIdx.x;
  if (idx >= Kpad*N) return;
  int n = idx / Kpad, k = idx % Kpad;
  float v = (k < K) ? src[(size_t)k*N + n] : 0.f;
  dst[(size_t)(rowOff + n)*Kpad + k] = (bf16_t)v;
}

__global__ void k_biascat(const float* __restrict__ bt, float* __restrict__ dst){
  int i = blockIdx.x*blockDim.x + threadIdx.x;
  if (i >= 768) return;
  dst[i] = (i < 512) ? 0.f : bt[i - 512];
}

// ---------------- bf16 WMMA GEMM: C = act(A[M x lda] @ B + bias) ---------------------------
// B given pre-transposed: Bt [Ndim x ldbt] with ldbt == Kdim (contiguous along K).
// block = 256 thr (8 waves); 64x128 block tile; wave w -> 32x32 sub-tile; double-buffered LDS
__global__ __launch_bounds__(256) void k_gemm(
    const bf16_t* __restrict__ A, int lda,
    const bf16_t* __restrict__ Bt, int ldbt,
    const float* __restrict__ bias,
    float* __restrict__ Cf, bf16_t* __restrict__ Cb, int ldc,
    int Kdim, int act)
{
  __shared__ __align__(16) bf16_t sA[2][BM*BK];
  __shared__ __align__(16) bf16_t sB[2][BN*BK];
  const int m0 = blockIdx.x * BM;
  const int n0 = blockIdx.y * BN;
  const int t = threadIdx.x;
  const int wave = t >> 5, lane = t & 31;
  const int half = lane >> 4, l15 = lane & 15;
  const int wm = (wave & 1) * 32;
  const int wn = (wave >> 1) * 32;
  // staging: A 64x32 -> 4 thr/row (1x b128 each); B 128x32 -> 2 thr/row (2x b128 each)
  const int ar = t >> 2, ac = (t & 3) * 8;
  const int br = t >> 1, bc = (t & 1) * 16;

  auto stage = [&](int buf, int k0){
    const bf16_t* ap = A  + (size_t)(m0 + ar)*lda  + k0 + ac;
    const bf16_t* bp = Bt + (size_t)(n0 + br)*ldbt + k0 + bc;
    *(v8bf*)&sA[buf][ar*BK + ac]     = *(const v8bf*)ap;
    *(v8bf*)&sB[buf][br*BK + bc]     = *(const v8bf*)bp;
    *(v8bf*)&sB[buf][br*BK + bc + 8] = *(const v8bf*)(bp + 8);
    if (k0 + 2*BK < Kdim){                      // pull tiles two steps ahead into cache
      __builtin_prefetch((const void*)(ap + 2*BK), 0, 3);
      __builtin_prefetch((const void*)(bp + 2*BK), 0, 3);
    }
  };

  v8f a00 = {}, a01 = {}, a10 = {}, a11 = {};
  stage(0, 0);
  __syncthreads();
  const int nk = Kdim >> 5;
  for (int kt = 0; kt < nk; ++kt){
    const int cur = kt & 1;
    if (kt + 1 < nk) stage(cur ^ 1, (kt + 1) << 5);
    v16bf fa0 = fragA(&sA[cur][(wm + l15)*BK],      half);
    v16bf fa1 = fragA(&sA[cur][(wm + 16 + l15)*BK], half);
    v16bf fb0 = fragB(&sB[cur][(wn + l15)*BK],      half);
    v16bf fb1 = fragB(&sB[cur][(wn + 16 + l15)*BK], half);
    a00 = wmma_bf16(fa0, fb0, a00);
    a01 = wmma_bf16(fa0, fb1, a01);
    a10 = wmma_bf16(fa1, fb0, a10);
    a11 = wmma_bf16(fa1, fb1, a11);
    __syncthreads();
  }
  #pragma unroll
  for (int mi = 0; mi < 2; ++mi){
    #pragma unroll
    for (int j = 0; j < 8; ++j){
      int row = m0 + wm + mi*16 + 8*half + j;
      #pragma unroll
      for (int ni = 0; ni < 2; ++ni){
        int col = n0 + wn + ni*16 + l15;
        float v = mi ? (ni ? a11[j] : a10[j]) : (ni ? a01[j] : a00[j]);
        if (bias) v += bias[col];
        if (act == 1) v = seluf(v);
        if (Cf) Cf[(size_t)row*ldc + col] = v;
        if (Cb) Cb[(size_t)row*ldc + col] = (bf16_t)v;
      }
    }
  }
}

// ---------------- LSH binning: argmax over [emb@rot[:, :25], -same] ------------------------
__global__ void k_binidx(const float* __restrict__ emb, const float* __restrict__ rot,
                         int* __restrict__ bidx){
  int r = blockIdx.x*blockDim.x + threadIdx.x;
  if (r >= ROWS) return;
  const float* e = emb + (size_t)r*DH;
  float mul[25];
  for (int j = 0; j < 25; ++j){
    float s = 0.f;
    for (int k = 0; k < DH; ++k) s += e[k] * rot[k*100 + j];
    mul[j] = s;
  }
  float best = -3.4e38f; int bi = 0;
  for (int j = 0; j < 50; ++j){
    float v = (j < 25) ? mul[j] : -mul[j - 25];
    if (v > best){ best = v; bi = j; }   // strict >: first max wins (argmax semantics)
  }
  bidx[r] = bi;
}

// ---------------- stable counting sort (matches stable argsort of bin ids) -----------------
__global__ void k_sortbins(const int* __restrict__ bidx, int* __restrict__ split){
  int w = blockIdx.x*blockDim.x + threadIdx.x;
  if (w >= BATCH*NBINS) return;
  int batch = w / NBINS, bin = w % NBINS;
  const int* bp = bidx + batch*NPTS;
  int start = 0;
  for (int i = 0; i < NPTS; ++i) start += (bp[i] < bin);
  int pos = start;
  int* out = split + batch*NPTS;
  for (int i = 0; i < NPTS; ++i) if (bp[i] == bin) out[pos++] = i;
}

// ---------------- per-chunk pairwise dist + top-16 + degree norm ---------------------------
__global__ __launch_bounds__(128) void k_edges(
    const float* __restrict__ emb, const int* __restrict__ split,
    float* __restrict__ vals, int* __restrict__ dstn, float* __restrict__ normv)
{
  __shared__ __align__(16) bf16_t parts[BINSZ*264]; // padded stride vs bank conflicts
  __shared__ float na[BINSZ];
  __shared__ int ids[BINSZ];
  const int blk = blockIdx.x;
  const int batch = blk / NBINS, chunk = blk % NBINS;
  const int base = batch * NPTS;
  const int t = threadIdx.x;
  if (t < BINSZ) ids[t] = split[base + chunk*BINSZ + t];
  __syncthreads();
  for (int e = t; e < BINSZ*DH; e += 128){
    int row = e >> 8, k = e & 255;
    parts[row*264 + k] = (bf16_t)emb[(size_t)(base + ids[row])*DH + k];
  }
  __syncthreads();
  if (t < BINSZ){
    float s = 0.f;
    for (int k = 0; k < DH; ++k){ float v = (float)parts[t*264 + k]; s += v*v; }
    na[t] = s;
  }
  __syncthreads();
  if (t >= BINSZ) return;
  const int i = t;
  float dmrow[BINSZ];
  for (int j = 0; j < BINSZ; ++j){
    float dot = 0.f;
    for (int k = 0; k < DH; ++k) dot += (float)parts[i*264 + k] * (float)parts[j*264 + k];
    float d2 = na[i] + na[j] - 2.f*dot;
    dmrow[j] = expf(-0.1f * sqrtf(fmaxf(d2, 1e-6f)));
  }
  unsigned long long m0 = 0ULL, m1 = 0ULL;
  float deg = 0.f;
  const int node = ids[i];
  for (int kk = 0; kk < KNBR; ++kk){
    float bv = -3.4e38f; int bj = 0;
    for (int j = 0; j < BINSZ; ++j){
      bool tk = (j < 64) ? ((m0 >> j) & 1ULL) : ((m1 >> (j - 64)) & 1ULL);
      if (!tk && dmrow[j] > bv){ bv = dmrow[j]; bj = j; } // strict >: tie -> lowest index
    }
    if (bj < 64) m0 |= 1ULL << bj; else m1 |= 1ULL << (bj - 64);
    vals[(size_t)(base + node)*KNBR + kk] = bv;
    dstn[(size_t)(base + node)*KNBR + kk] = ids[bj];
    deg += bv;
  }
  normv[base + node] = 1.f / sqrtf(deg + 1e-6f);
}

// ---------------- gated aggregation: h = selu(g*agg + (1-g)*fhet) --------------------------
// fused [ROWS x 768] = [x@theta | x@Wh | x@Wt + bt]
__global__ __launch_bounds__(256) void k_combine(
    const float* __restrict__ fused, const float* __restrict__ vals,
    const int* __restrict__ dstn, const float* __restrict__ normv,
    bf16_t* __restrict__ hout)
{
  const int r = blockIdx.x;
  const int base = (r / NPTS) * NPTS;
  const int t = threadIdx.x;
  __shared__ float wk[KNBR];
  __shared__ size_t drow[KNBR];
  __shared__ float nself;
  if (t < KNBR){
    float v = vals[(size_t)r*KNBR + t];
    int d = dstn[(size_t)r*KNBR + t];
    wk[t] = v * normv[base + d];
    drow[t] = (size_t)(base + d) * 768;
  }
  if (t == 0) nself = normv[r];
  __syncthreads();
  const int c = t;
  float agg = 0.f;
  #pragma unroll
  for (int k = 0; k < KNBR; ++k) agg += wk[k] * fused[drow[k] + c];
  agg *= nself;
  float fhet = fused[(size_t)r*768 + 256 + c];
  float gp   = fused[(size_t)r*768 + 512 + c];
  float g = 1.f / (1.f + expf(-gp));
  hout[(size_t)r*DH + c] = (bf16_t)seluf(g*agg + (1.f - g)*fhet);
}

// ---------------- skinny output GEMM (N=8 or 4) --------------------------------------------
__global__ void k_out(const bf16_t* __restrict__ y, const float* __restrict__ W,
                      const float* __restrict__ b, float* __restrict__ out, int nout)
{
  int idx = blockIdx.x*blockDim.x + threadIdx.x;
  if (idx >= ROWS*nout) return;
  int r = idx / nout, oc = idx % nout;
  const bf16_t* yr = y + (size_t)r*DH;
  float s = b[oc];
  for (int c = 0; c < DH; ++c) s += (float)yr[c] * W[c*nout + oc];
  out[idx] = s;
}

extern "C" void kernel_launch(void* const* d_in, const int* in_sizes, int n_in,
                              void* d_out, int out_size, void* d_ws, size_t ws_size,
                              hipStream_t stream)
{
  (void)in_sizes; (void)n_in; (void)out_size; (void)ws_size;
  const float* X     = (const float*)d_in[0];
  const float* rot   = (const float*)d_in[1];
  const float* W_emb = (const float*)d_in[2];
  const float* b_emb = (const float*)d_in[3];
  const float* Wenc[2] = {(const float*)d_in[4],  (const float*)d_in[14]};
  const float* benc[2] = {(const float*)d_in[5],  (const float*)d_in[15]};
  const float* th[2]   = {(const float*)d_in[6],  (const float*)d_in[16]};
  const float* Wh[2]   = {(const float*)d_in[7],  (const float*)d_in[17]};
  const float* Wt[2]   = {(const float*)d_in[8],  (const float*)d_in[18]};
  const float* bt[2]   = {(const float*)d_in[9],  (const float*)d_in[19]};
  const float* Wdec[2] = {(const float*)d_in[10], (const float*)d_in[20]};
  const float* bdec[2] = {(const float*)d_in[11], (const float*)d_in[21]};
  const float* Wout[2] = {(const float*)d_in[12], (const float*)d_in[22]};
  const float* bout[2] = {(const float*)d_in[13], (const float*)d_in[23]};
  const int nouts[2] = {8, 4};

  char* wsb = (char*)d_ws;
  size_t off = 0;
  auto alloc = [&](size_t bytes) -> char* {
    off = (off + 255) & ~(size_t)255;
    char* p = wsb + off; off += bytes; return p;
  };
  bf16_t* enc     = (bf16_t*)alloc((size_t)ROWS*32*2);
  bf16_t* pwT_emb = (bf16_t*)alloc(256*32*2);                    // [256 n][32 k]
  bf16_t* pwT_enc[2] = {(bf16_t*)alloc(256*32*2),  (bf16_t*)alloc(256*32*2)};
  bf16_t* pw3T[2]    = {(bf16_t*)alloc(768*256*2), (bf16_t*)alloc(768*256*2)}; // [768 n][256 k]
  bf16_t* pwT_dec[2] = {(bf16_t*)alloc(256*256*2), (bf16_t*)alloc(256*256*2)};
  float*  bcat[2]    = {(float*)alloc(768*4),      (float*)alloc(768*4)};
  int*    bidx  = (int*)alloc((size_t)ROWS*4);
  int*    split = (int*)alloc((size_t)ROWS*4);
  float*  vals  = (float*)alloc((size_t)ROWS*KNBR*4);
  int*    dstn  = (int*)alloc((size_t)ROWS*KNBR*4);
  float*  normv = (float*)alloc((size_t)ROWS*4);
  bf16_t* xb = (bf16_t*)alloc((size_t)ROWS*DH*2);
  bf16_t* hb = (bf16_t*)alloc((size_t)ROWS*DH*2);
  bf16_t* yb = (bf16_t*)alloc((size_t)ROWS*DH*2);
  // big f32 region: emb [ROWS x 256] (phase A/B) and fused [ROWS x 768] (phase C) share it
  float* big = (float*)alloc((size_t)ROWS*768*4);
  float* emb = big;
  float* fused = big;

  const int TPB = 256;
  k_packT<<<(32*256 + TPB-1)/TPB, TPB, 0, stream>>>(W_emb, pwT_emb, 26, 256, 32, 0);
  for (int br = 0; br < 2; ++br){
    k_packT<<<(32*256  + TPB-1)/TPB, TPB, 0, stream>>>(Wenc[br], pwT_enc[br], 26, 256, 32, 0);
    k_packT<<<(256*256 + TPB-1)/TPB, TPB, 0, stream>>>(th[br],   pw3T[br], 256, 256, 256, 0);
    k_packT<<<(256*256 + TPB-1)/TPB, TPB, 0, stream>>>(Wh[br],   pw3T[br], 256, 256, 256, 256);
    k_packT<<<(256*256 + TPB-1)/TPB, TPB, 0, stream>>>(Wt[br],   pw3T[br], 256, 256, 256, 512);
    k_packT<<<(256*256 + TPB-1)/TPB, TPB, 0, stream>>>(Wdec[br], pwT_dec[br], 256, 256, 256, 0);
    k_biascat<<<3, TPB, 0, stream>>>(bt[br], bcat[br]);
  }
  k_encode<<<(ROWS*32 + TPB-1)/TPB, TPB, 0, stream>>>(X, enc);
  // emb = enc @ W_emb + b_emb (f32 out, used by binning + pairwise)
  k_gemm<<<dim3(ROWS/BM, 256/BN), TPB, 0, stream>>>(enc, 32, pwT_emb, 32, b_emb,
                                                    emb, nullptr, 256, 32, 0);
  k_binidx<<<(ROWS + TPB-1)/TPB, TPB, 0, stream>>>(emb, rot, bidx);
  k_sortbins<<<(BATCH*NBINS + TPB-1)/TPB, TPB, 0, stream>>>(bidx, split);
  k_edges<<<BATCH*NBINS, 128, 0, stream>>>(emb, split, vals, dstn, normv);

  float* outp = (float*)d_out;
  for (int br = 0; br < 2; ++br){
    // x = selu(enc @ Wenc + benc)  (bf16 out only)
    k_gemm<<<dim3(ROWS/BM, 256/BN), TPB, 0, stream>>>(enc, 32, pwT_enc[br], 32, benc[br],
                                                      nullptr, xb, 256, 32, 1);
    // fused = x @ [theta | Wh | Wt] + [0|0|bt]  (f32 out, N=768)
    k_gemm<<<dim3(ROWS/BM, 768/BN), TPB, 0, stream>>>(xb, 256, pw3T[br], 256, bcat[br],
                                                      fused, nullptr, 768, 256, 0);
    k_combine<<<ROWS, TPB, 0, stream>>>(fused, vals, dstn, normv, hb);
    // y = selu(h @ Wdec + bdec)  (bf16 out only)
    k_gemm<<<dim3(ROWS/BM, 256/BN), TPB, 0, stream>>>(hb, 256, pwT_dec[br], 256, bdec[br],
                                                      nullptr, yb, 256, 256, 1);
    int nout = nouts[br];
    k_out<<<(ROWS*nout + TPB-1)/TPB, TPB, 0, stream>>>(yb, Wout[br], bout[br], outp, nout);
    outp += (size_t)ROWS * nout;
  }
}
